// SynthesisLayer_46042049413340
// MI455X (gfx1250) — matrix operand order repaired and checked by reading, not compiled, not run on hardware
//
#include <hip/hip_runtime.h>
#include <hip/hip_bf16.h>

// ---------------- types ----------------
typedef __attribute__((ext_vector_type(16))) __bf16 v16bf;
typedef __attribute__((ext_vector_type(8)))  __bf16 v8bf;
typedef __attribute__((ext_vector_type(8)))  float  v8f;

// ---------------- problem constants ----------------
constexpr int NB   = 16;
constexpr int CIN  = 512;
constexpr int COUT = 512;
constexpr int WDIM = 512;
constexpr int RES  = 64;
constexpr int HC   = 66;           // conv output spatial (64 + 2*2 - 3 + 1)
constexpr int NPIX = HC * HC;      // 4356
constexpr int PADW = 68;           // zero-padded activation row (64 + 2*2)
constexpr int PADPIX = PADW * PADW;// 4624
constexpr int TAPS = 12;
constexpr int HUP  = 138;          // upsampled spatial extent
constexpr int KTOT = CIN * 9;      // 4608
constexpr int STAGES = KTOT / 64;  // 72 LDS stages of K=64

__device__ __forceinline__ unsigned short f2bf_bits(float v) {
    __hip_bfloat16 h = __float2bfloat16(v);
    return *reinterpret_cast<unsigned short*>(&h);
}

__device__ __forceinline__ v16bf ldfragA(const unsigned short* base, int half) {
    // A (16x32 bf16): lane = row(M); elems 0..15 = K half*16+0..15
    v8bf lo = *(const v8bf*)(base + half * 16);
    v8bf hi = *(const v8bf*)(base + half * 16 + 8);
    return __builtin_shufflevector(lo, hi, 0,1,2,3,4,5,6,7,8,9,10,11,12,13,14,15);
}
__device__ __forceinline__ v16bf ldfragB(const unsigned short* base) {
    // B (32x16 bf16): lane = col(N); lanes 0-15 hold K 0..15, lanes 16-31 hold K 16..31
    v8bf lo = *(const v8bf*)(base);
    v8bf hi = *(const v8bf*)(base + 8);
    return __builtin_shufflevector(lo, hi, 0,1,2,3,4,5,6,7,8,9,10,11,12,13,14,15);
}
__device__ __forceinline__ v8f wmma_bf16(v16bf a, v16bf b, v8f c) {
    return __builtin_amdgcn_wmma_f32_16x16x32_bf16(false, a, false, b, (short)0, c, false, false);
}

// CDNA5 async global->LDS copy (ISA 15.18.3 op 98, ASYNCcnt) + drain
__device__ __forceinline__ void async_copy_b128(unsigned lds_off, const void* gptr) {
    asm volatile("global_load_async_to_lds_b128 %0, %1, off"
                 :: "v"(lds_off), "v"((unsigned long long)(size_t)gptr) : "memory");
}
__device__ __forceinline__ void wait_async0() {
    asm volatile("s_wait_asynccnt 0" ::: "memory");
}

// ---------------- stage 1: styles = w @ A^T / sqrt(512) + b ----------------
__global__ __launch_bounds__(256) void k_styles(const float* __restrict__ w,
                                                const float* __restrict__ aw,
                                                const float* __restrict__ ab,
                                                float* __restrict__ styles) {
    int n = blockIdx.y;
    int c = blockIdx.x * 256 + threadIdx.x;
    const float* wr = w + n * WDIM;
    const float* ar = aw + c * WDIM;
    float acc = 0.f;
    for (int k = 0; k < WDIM; ++k) acc += wr[k] * ar[k];
    styles[n * CIN + c] = acc * 0.04419417382415922f + ab[c];
}

// ---------------- stage 2: snf = rsqrt(mean(styles^2)) ----------------
__global__ __launch_bounds__(256) void k_style_norm(const float* __restrict__ styles,
                                                    float* __restrict__ snf) {
    __shared__ float red[256];
    float acc = 0.f;
    for (int i = threadIdx.x; i < NB * CIN; i += 256) { float v = styles[i]; acc += v * v; }
    red[threadIdx.x] = acc; __syncthreads();
    for (int s = 128; s > 0; s >>= 1) {
        if (threadIdx.x < (unsigned)s) red[threadIdx.x] += red[threadIdx.x + s];
        __syncthreads();
    }
    if (threadIdx.x == 0) *snf = rsqrtf(red[0] / (float)(NB * CIN));
}

// ---------------- stage 3: per-out-channel weight rms ----------------
__global__ __launch_bounds__(256) void k_wnorm(const float* __restrict__ cw,
                                               float* __restrict__ r) {
    __shared__ float red[256];
    int o = blockIdx.x;
    const float* p = cw + (size_t)o * CIN * 9;
    float acc = 0.f;
    for (int i = threadIdx.x; i < CIN * 9; i += 256) { float v = p[i]; acc += v * v; }
    red[threadIdx.x] = acc; __syncthreads();
    for (int s = 128; s > 0; s >>= 1) {
        if (threadIdx.x < (unsigned)s) red[threadIdx.x] += red[threadIdx.x + s];
        __syncthreads();
    }
    if (threadIdx.x == 0) r[o] = rsqrtf(red[0] / (float)(CIN * 9));
}

// ---------------- stage 4: wt(bf16, [o][tap][i]) + q[o][i] = sum_tap wt^2 ----------------
__global__ __launch_bounds__(256) void k_wtb_q(const float* __restrict__ cw,
                                               const float* __restrict__ r,
                                               unsigned short* __restrict__ wtb,
                                               float* __restrict__ q) {
    int idx = blockIdx.x * 256 + threadIdx.x;  // o*512 + i
    int o = idx >> 9, i = idx & 511;
    float ro = r[o];
    const float* p = cw + ((size_t)o * CIN + i) * 9;
    unsigned short* wb = wtb + (size_t)o * (CIN * 9);
    float s = 0.f;
#pragma unroll
    for (int t = 0; t < 9; ++t) {
        float v = p[t] * ro;
        s += v * v;
        wb[t * CIN + i] = f2bf_bits(v);
    }
    q[idx] = s;
}

// ---------------- stage 5: demod d[n][o] ----------------
__global__ __launch_bounds__(256) void k_demod(const float* __restrict__ styles,
                                               const float* __restrict__ snf,
                                               const float* __restrict__ q,
                                               float* __restrict__ dcoef) {
    int n = blockIdx.y;
    int o = blockIdx.x * 256 + threadIdx.x;
    float sn = *snf;
    const float* sr = styles + n * CIN;
    const float* qr = q + (size_t)o * CIN;
    float acc = 0.f;
    for (int i = 0; i < CIN; ++i) { float sv = sr[i] * sn; acc += sv * sv * qr[i]; }
    dcoef[n * COUT + o] = rsqrtf(acc + 1e-8f);
}

// ---------------- stage 6: xs(bf16, zero-padded 68x68) = x * s[n,i] ----------------
__global__ __launch_bounds__(256) void k_modx(const float* __restrict__ x,
                                              const float* __restrict__ styles,
                                              const float* __restrict__ snf,
                                              unsigned short* __restrict__ xp) {
    int ci = blockIdx.y;                        // n*512 + i
    int e  = blockIdx.x * 256 + threadIdx.x;    // padded pixel
    if (e >= PADPIX) return;
    int h = e / PADW, ww = e % PADW;
    float v = 0.f;
    if (h >= 2 && h < 66 && ww >= 2 && ww < 66)
        v = x[(size_t)ci * (RES * RES) + (h - 2) * RES + (ww - 2)] * styles[ci] * (*snf);
    xp[(size_t)ci * PADPIX + e] = f2bf_bits(v);
}

// ---------------- stage 7: implicit-GEMM conv via WMMA bf16 ----------------
// Block tile 128(O) x 64(P), K=64 per LDS stage (8 WMMAs / wave / barrier).
// Weights: async global->LDS (ASYNCcnt); activations: padded gather via registers.
__global__ __launch_bounds__(256) void k_conv_wmma(const unsigned short* __restrict__ xp,
                                                   const unsigned short* __restrict__ wtb,
                                                   const float* __restrict__ dcoef,
                                                   const float* __restrict__ cbias,
                                                   float* __restrict__ y) {
    __shared__ unsigned short lW[2][128][64];   // [buf][o][k]  32KB
    __shared__ unsigned short lX[2][64][64];    // [buf][p][k]  16KB (transposed for B frags)

    const int n  = blockIdx.z;
    const int o0 = blockIdx.y * 128;
    const int p0 = blockIdx.x * 64;
    const int tid  = threadIdx.x;
    const int lane = tid & 31;
    const int wave = tid >> 5;   // 0..7
    const int wo   = wave >> 1;  // 0..3  -> 32-row M quadrant
    const int wp   = wave & 1;   // 0..1  -> 32-col N quadrant
    const int row  = lane & 15;
    const int half = lane >> 4;

    // W-loader: thread covers o-row tid>>1, 32 contiguous k at (tid&1)*32 (64B = 4x b128 async)
    const int wrow = tid >> 1;
    const int wkc  = (tid & 1) * 32;
    // X-loader: thread covers channel tid>>2, 16 consecutive pixels at (tid&3)*16
    const int xch  = tid >> 2;
    const int xpg  = (tid & 3) * 16;

    int pix[16];
#pragma unroll
    for (int e = 0; e < 16; ++e) {
        int p = p0 + xpg + e;
        if (p >= NPIX) p = NPIX - 1;           // clamped; stores are guarded
        pix[e] = (p / HC) * PADW + (p % HC);   // padded base offset (ky=kx=0)
    }
    const unsigned short* xbase = xp + (size_t)n * CIN * PADPIX;
    const unsigned short* wbase = wtb + (size_t)(o0 + wrow) * KTOT + wkc;

    v8f c00 = {}, c01 = {}, c10 = {}, c11 = {};
    unsigned short xr[16];

    // stage s covers taps r = s>>3 (same for both halves), channels ci = (s&7)*64
    auto issueW = [&](int b, int s) {
        int r = s >> 3, ci = (s & 7) << 6;
        const unsigned short* g = wbase + r * CIN + ci;
        unsigned l = (unsigned)(size_t)&lW[b][wrow][wkc];
        async_copy_b128(l,      g);
        async_copy_b128(l + 16, g + 8);
        async_copy_b128(l + 32, g + 16);
        async_copy_b128(l + 48, g + 24);
    };
    auto loadX = [&](int s) {
        int r = s >> 3, ci = (s & 7) << 6;
        int ky = r / 3, kx = r - ky * 3;
        int koff = ky * PADW + kx;
        const unsigned short* xc = xbase + (size_t)(ci + xch) * PADPIX;
#pragma unroll
        for (int e = 0; e < 16; ++e) xr[e] = xc[pix[e] + koff];
    };
    auto storeX = [&](int b) {
#pragma unroll
        for (int e = 0; e < 16; ++e) lX[b][xpg + e][xch] = xr[e];
    };

    // prologue: stage 0 into buffer 0
    issueW(0, 0);
    loadX(0);
    storeX(0);
    wait_async0();
    __syncthreads();

    for (int s = 0; s < STAGES; ++s) {
        const int cur = s & 1, nxt = cur ^ 1;
        if (s + 1 < STAGES) {
            issueW(nxt, s + 1);
            loadX(s + 1);
            int ci2 = ((s + 2) & 7) << 6;      // warm L2 on the activation stream
            __builtin_prefetch(xbase + (size_t)(ci2 + xch) * PADPIX + pix[0], 0, 0);
        }
        // 2 sub-steps of K=32: 2x2 outer-product register block, 4 WMMAs each
#pragma unroll
        for (int sub = 0; sub < 2; ++sub) {
            const int ko = sub * 32;
            v16bf a0 = ldfragA(&lW[cur][wo * 32 + row][ko],      half);
            v16bf a1 = ldfragA(&lW[cur][wo * 32 + 16 + row][ko], half);
            v16bf b0 = ldfragB(&lX[cur][wp * 32 + row][ko + half * 16]);
            v16bf b1 = ldfragB(&lX[cur][wp * 32 + 16 + row][ko + half * 16]);
            c00 = wmma_bf16(a0, b0, c00);
            c01 = wmma_bf16(a0, b1, c01);
            c10 = wmma_bf16(a1, b0, c10);
            c11 = wmma_bf16(a1, b1, c11);
        }
        if (s + 1 < STAGES) storeX(nxt);
        wait_async0();
        __syncthreads();
    }

    // store: D layout -- VGPR g: M=g (lanes 0-15) / M=g+8 (lanes 16-31); N = lane%16
    float* ybase = y + (size_t)n * COUT * NPIX;
    const float* dbase = dcoef + n * COUT;
#pragma unroll
    for (int ot = 0; ot < 2; ++ot) {
#pragma unroll
        for (int pt = 0; pt < 2; ++pt) {
            v8f c = (ot == 0) ? (pt == 0 ? c00 : c01) : (pt == 0 ? c10 : c11);
            int ob = o0 + wo * 32 + ot * 16 + half * 8;
            int p  = p0 + wp * 32 + pt * 16 + row;
            if (p < NPIX) {
#pragma unroll
                for (int g = 0; g < 8; ++g) {
                    int o = ob + g;
                    ybase[(size_t)o * NPIX + p] = c[g] * dbase[o] + cbias[o];
                }
            }
        }
    }
}

// ---------------- stage 8: filtered_lrelu (one block per (n,c) image) ----------------
// Polyphase: up-by-2 FIR touches only 6 of 12 taps per output phase.
__global__ __launch_bounds__(256) void k_flrelu(const float* __restrict__ y,
                                                const float* __restrict__ fup,
                                                const float* __restrict__ fdn,
                                                float* __restrict__ out) {
    __shared__ float buf[13524];      // t1[138*66]=9108 | in[66*66]=4356 (aliased by t2[138*32]=4416)
    __shared__ float f0[TAPS], f1[TAPS];
    float* t1 = buf;
    float* in = buf + 9108;
    float* t2 = buf + 9108;           // reuses `in` region after stage 1

    const int tid = threadIdx.x;
    const int img = blockIdx.x;       // n*512 + c
    const float* src = y + (size_t)img * NPIX;
    float* dst = out + (size_t)img * RES * RES;

    if (tid < TAPS) { f0[tid] = fup[tid]; f1[tid] = fdn[tid]; }
    for (int i = tid; i < NPIX; i += 256) in[i] = src[i];
    __syncthreads();

    // vertical up-FIR (polyphase 6-tap): t1[r][c], r<138, c<66
    for (int e = tid; e < HUP * HC; e += 256) {
        int r = e / HC, c = e % HC;
        int t0 = (r + 1) & 1;                     // taps with (r+t-9) even
        float acc = 0.f;
#pragma unroll
        for (int tt = 0; tt < 6; ++tt) {
            int t = t0 + 2 * tt;
            int qy = (r + t - 9) >> 1;
            if ((unsigned)qy < (unsigned)HC) acc += f0[t] * in[qy * HC + c];
        }
        t1[e] = acc;
    }
    __syncthreads();

    // two 32-column halves: (horiz up + lrelu + horiz down) then vertical down
    for (int h = 0; h < 2; ++h) {
        for (int e = tid; e < HUP * 32; e += 256) {
            int r = e >> 5, jj = e & 31;
            int j = h * 32 + jj;
            const float* rowp = t1 + r * HC;
            float acc = 0.f;
#pragma unroll
            for (int tt = 0; tt < TAPS; ++tt) {
                int idx = 2 * j + tt;             // upsampled x position
                int t0u = (tt + 1) & 1;           // taps with (idx+t-9) even
                float u = 0.f;
#pragma unroll
                for (int t2i = 0; t2i < 6; ++t2i) {
                    int t = t0u + 2 * t2i;
                    int qx = (idx + t - 9) >> 1;
                    if ((unsigned)qx < (unsigned)HC) u += f0[t] * rowp[qx];
                }
                u *= 4.0f;                        // up gain = up^2
                u = (u >= 0.f ? u : 0.2f * u) * 1.4142135623730951f;
                u = fminf(fmaxf(u, -256.f), 256.f);
                acc += f1[tt] * u;
            }
            t2[e] = acc;
        }
        __syncthreads();
        // vertical down-FIR: out[i][h*32+jj]
        for (int e = tid; e < RES * 32; e += 256) {
            int i = e >> 5, jj = e & 31;
            float acc = 0.f;
#pragma unroll
            for (int t = 0; t < TAPS; ++t)
                acc += f1[t] * t2[(2 * i + t) * 32 + jj];
            dst[i * RES + h * 32 + jj] = acc;
        }
        __syncthreads();
    }
}

// ---------------- launch ----------------
extern "C" void kernel_launch(void* const* d_in, const int* in_sizes, int n_in,
                              void* d_out, int out_size, void* d_ws, size_t ws_size,
                              hipStream_t stream) {
    const float* x  = (const float*)d_in[0];
    const float* w  = (const float*)d_in[1];
    const float* aw = (const float*)d_in[2];
    const float* ab = (const float*)d_in[3];
    const float* cw = (const float*)d_in[4];
    const float* cb = (const float*)d_in[5];
    const float* fu = (const float*)d_in[6];
    const float* fd = (const float*)d_in[7];
    float* out = (float*)d_out;

    char* ws = (char*)d_ws;
    size_t off = 0;
    auto alloc = [&](size_t bytes) -> void* {
        void* p = ws + off;
        off = (off + bytes + 255) & ~(size_t)255;
        return p;
    };
    float*          styles = (float*)alloc((size_t)NB * CIN * 4);
    float*          snf    = (float*)alloc(4);
    float*          rr     = (float*)alloc((size_t)COUT * 4);
    float*          q      = (float*)alloc((size_t)COUT * CIN * 4);
    float*          dco    = (float*)alloc((size_t)NB * COUT * 4);
    unsigned short* wtb    = (unsigned short*)alloc((size_t)COUT * KTOT * 2);
    unsigned short* xpb    = (unsigned short*)alloc((size_t)NB * CIN * PADPIX * 2);
    float*          ybuf   = (float*)alloc((size_t)NB * COUT * NPIX * 4);
    (void)ws_size; (void)in_sizes; (void)n_in; (void)out_size;

    k_styles    <<<dim3(2, NB), 256, 0, stream>>>(w, aw, ab, styles);
    k_style_norm<<<1, 256, 0, stream>>>(styles, snf);
    k_wnorm     <<<COUT, 256, 0, stream>>>(cw, rr);
    k_wtb_q     <<<(COUT * CIN) / 256, 256, 0, stream>>>(cw, rr, wtb, q);
    k_demod     <<<dim3(2, NB), 256, 0, stream>>>(styles, snf, q, dco);
    k_modx      <<<dim3((PADPIX + 255) / 256, NB * CIN), 256, 0, stream>>>(x, styles, snf, xpb);
    k_conv_wmma <<<dim3((NPIX + 63) / 64, COUT / 128, NB), 256, 0, stream>>>(xpb, wtb, dco, cb, ybuf);
    k_flrelu    <<<NB * COUT, 256, 0, stream>>>(ybuf, fu, fd, out);
}
